// StructuredLayer_14396730377042
// MI455X (gfx1250) — matrix operand order repaired and checked by reading, not compiled, not run on hardware
//
#include <hip/hip_runtime.h>
#include <hip/hip_bf16.h>

// ---------------------------------------------------------------------------
// Structured-sparse linear layer for MI455X (gfx1250, wave32, WMMA).
//
//   out[n,o] = sum_k x[n,k] * (weight*mask*gauge)[o,k] + bias[o]
//
// Output o uses 81 contiguous (mod 4096) input columns starting at 81*o, so a
// 16-output group g needs one 1296-wide column window at (1296*g) & 4095.
// Pipeline:
//   K0: x f32 -> bf16, rows padded to 4112 (first 16 cols replicated).
//   K1: weight*mask*gauge -> bf16 in exact WMMA B-fragment order.
//   K2: banded GEMM, v_wmma_f32_16x16x32_bf16, K padded to 42 tiles of 32.
//       Band wrap is applied ONLY to the uniform tile base (SALU), never to
//       per-lane offsets: tilebase is a multiple of 16, so all per-lane
//       accesses land in [tilebase, tilebase+31] <= 4111, inside row padding.
//       Each wave computes TWO 16x16 C tiles sharing each B fragment.
// x_bf16 (67MB) + packed W (11MB) stay L2-resident (192MB) across all 256
// group passes; out is written with non-temporal stores.
// ---------------------------------------------------------------------------

#define IN_F   4096
#define OUT_F  4096
#define NTOK   8192
#define GROUPS 256            // OUT_F / 16
#define KWIN   1296           // 16 * 81 columns per group window
#define KTILES 42             // 1296 padded to 1344 = 42 * 32
#define XPAD   4112           // padded bf16 row length (4096 + 16)

typedef __attribute__((ext_vector_type(16))) __bf16        v16bf;
typedef __attribute__((ext_vector_type(8)))  float         v8f;
typedef __attribute__((ext_vector_type(8)))  unsigned int  v8u;

// float -> bf16 bits, round-to-nearest-even.
__device__ __forceinline__ unsigned short f2bf_bits(float f) {
  unsigned int u = __builtin_bit_cast(unsigned int, f);
  u = (u + 0x7FFFu + ((u >> 16) & 1u)) >> 16;
  return (unsigned short)u;
}

__device__ __forceinline__ v16bf load_a_frag(const unsigned short* p) {
  const uint4 lo = *(const uint4*)(p);
  const uint4 hi = *(const uint4*)(p + 16);
  v8u au;
  au[0] = lo.x; au[1] = lo.y; au[2] = lo.z; au[3] = lo.w;
  au[4] = hi.x; au[5] = hi.y; au[6] = hi.z; au[7] = hi.w;
  return __builtin_bit_cast(v16bf, au);
}

// ---------------------------------------------------------------------------
// Kernel 0: x[8192][4096] f32 -> xb[8192][4112] bf16, cols 4096..4111 mirror
// cols 0..15 so band-window tiles never wrap inside a fragment load.
// ---------------------------------------------------------------------------
__global__ __launch_bounds__(256)
void cvt_x_kernel(const float* __restrict__ x, unsigned short* __restrict__ xb) {
  const int n  = blockIdx.y;
  const int c4 = (blockIdx.x * 256 + threadIdx.x) * 4;   // 4 elems / thread
  if (c4 >= XPAD) return;
  // c4 and 4096 are multiples of 4: the 4-run never straddles the wrap point.
  const float4 v = *(const float4*)(x + (size_t)n * IN_F + (c4 & (IN_F - 1)));
  unsigned long long p =  (unsigned long long)f2bf_bits(v.x)
                       | ((unsigned long long)f2bf_bits(v.y) << 16)
                       | ((unsigned long long)f2bf_bits(v.z) << 32)
                       | ((unsigned long long)f2bf_bits(v.w) << 48);
  *(unsigned long long*)(xb + (size_t)n * XPAD + c4) = p;
}

// ---------------------------------------------------------------------------
// Kernel 1: pack W_eff = weight*mask*gauge into bf16 in the exact
// v_wmma_f32_16x16x32_bf16 B-fragment order:
//   elem ((g*KTILES + kt)*32 + lane)*16 + j
//     = B[k = (lane>>4)*8 + (j&7) + (j>>3)*16][n = lane&15]  for K-tile kt of group g
//     = W_eff[16g + (lane&15)][(1296g + 32kt + 8(lane>>4) + (j&7) + 16(j>>3)) & 4095].
// The mask zeroes kwin >= 1296 (padding never lands in this group's bands).
// ---------------------------------------------------------------------------
__global__ __launch_bounds__(256)
void pack_weights_kernel(const float* __restrict__ weight,
                         const float* __restrict__ mask,
                         const float* __restrict__ gauge,
                         unsigned short* __restrict__ wp) {
  const unsigned int idx  = blockIdx.x * 256u + threadIdx.x;  // < GROUPS*KTILES*512
  const unsigned int perg = KTILES * 512u;
  const unsigned int g    = idx / perg;
  const unsigned int r    = idx % perg;
  const unsigned int kt   = r >> 9;
  const unsigned int lane = (r >> 4) & 31u;
  const unsigned int j    = r & 15u;

  const unsigned int kwin = kt * 32u + ((lane >> 4) * 8u) + (j & 7u) + ((j >> 3) * 16u);   // 0..1343
  const unsigned int o    = g * 16u + (lane & 15u);
  const unsigned int c    = (KWIN * g + kwin) & (IN_F - 1);
  const size_t widx = (size_t)o * IN_F + c;

  wp[idx] = f2bf_bits(weight[widx] * mask[widx] * gauge[widx]);
}

// ---------------------------------------------------------------------------
// Kernel 2: banded GEMM.  grid = (NTOK/128, GROUPS), block = 128 (4 waves).
// Each wave: two 16-token x 16-output C tiles (32 tokens), 42 K-tiles.
// Per K-tile: 4x b128 A loads + 32B B load + 2 independent WMMAs.
// ---------------------------------------------------------------------------
__global__ __launch_bounds__(128)
void banded_gemm_kernel(const unsigned short* __restrict__ xb,
                        const unsigned short* __restrict__ wp,
                        const float* __restrict__ bias,
                        float* __restrict__ out) {
  const int lane = threadIdx.x & 31;
  const int wave = threadIdx.x >> 5;
  const int g    = blockIdx.y;
  const int mbase   = blockIdx.x * 128 + wave * 32;    // 32 tokens per wave
  const int colbase = (KWIN * g) & (IN_F - 1);         // uniform (SALU)

  const int li = lane & 15;
  const int hi = lane >> 4;                            // 0 or 1

  // ISA 16-bit A layout: lanes 0-15: M=lane,    K = {0..7, 16..23}
  //                      lanes 16-31: M=lane-16, K = {8..15, 24..31}
  // Loop-invariant per-lane element offsets for the two token tiles:
  const size_t rowoff0 = (size_t)(mbase + li) * XPAD + (hi * 8);
  const size_t rowoff1 = rowoff0 + (size_t)16 * XPAD;

  // Pre-packed B: one contiguous v16bf (32B, 32B-aligned) per lane per K tile.
  const v16bf* __restrict__ bptr =
      (const v16bf*)wp + (size_t)g * (KTILES * 32) + lane;

  v8f acc0 = {};
  v8f acc1 = {};
#pragma unroll 3
  for (int kt = 0; kt < KTILES; ++kt) {
    // Uniform, SALU-only band base with wrap; multiple of 16 => <= 4080, so
    // every per-lane access (<= tilebase+31 <= 4111) stays in the padded row.
    const int tilebase = (colbase + kt * 32) & (IN_F - 1);
    const unsigned short* __restrict__ pt = xb + tilebase;

    const v16bf a0 = load_a_frag(pt + rowoff0);
    const v16bf a1 = load_a_frag(pt + rowoff1);
    const v16bf b  = bptr[kt * 32];

    acc0 = __builtin_amdgcn_wmma_f32_16x16x32_bf16(
        false, a0, false, b, (short)0, acc0, false, false);
    acc1 = __builtin_amdgcn_wmma_f32_16x16x32_bf16(
        false, a1, false, b, (short)0, acc1, false, false);
  }

  // C/D layout: lane -> output col 16g+li, token rows M = hi*8 + r in VGPR r.
  // Dword stores are coalesced (lanes 0-15 = 64 consecutive bytes per row).
  const int ocol = g * 16 + li;
  const float bv = bias[ocol];
  float* __restrict__ op0 = out + (size_t)(mbase + hi * 8) * OUT_F + ocol;
  float* __restrict__ op1 = op0 + (size_t)16 * OUT_F;
#pragma unroll
  for (int r = 0; r < 8; ++r) {
    __builtin_nontemporal_store(acc0[r] + bv, op0 + (size_t)r * OUT_F);
    __builtin_nontemporal_store(acc1[r] + bv, op1 + (size_t)r * OUT_F);
  }
}

// ---------------------------------------------------------------------------
extern "C" void kernel_launch(void* const* d_in, const int* in_sizes, int n_in,
                              void* d_out, int out_size, void* d_ws, size_t ws_size,
                              hipStream_t stream) {
  const float* x      = (const float*)d_in[0];  // [8192,4096]
  const float* weight = (const float*)d_in[1];  // [4096,4096]
  const float* bias   = (const float*)d_in[2];  // [4096]
  const float* mask   = (const float*)d_in[3];  // [4096,4096]
  const float* gauge  = (const float*)d_in[4];  // [4096,4096]
  float* out = (float*)d_out;                   // [8192,4096]

  // Workspace layout (78.4 MB total):
  //   [0, 11.0MB)        packed W fragments (GROUPS*KTILES*512 bf16)
  //   [11.0MB, +67.4MB)  padded bf16 x (NTOK * XPAD)
  unsigned short* wp = (unsigned short*)d_ws;
  const size_t wp_elems = (size_t)GROUPS * KTILES * 512;      // 5,505,024
  unsigned short* xbf = wp + wp_elems;                        // 256B-aligned

  {  // K0: x -> padded bf16
    dim3 grid((XPAD / 4 + 255) / 256, NTOK);                  // (5, 8192)
    cvt_x_kernel<<<grid, 256, 0, stream>>>(x, xbf);
  }
  {  // K1: pack weights (exact multiple of 256 threads)
    pack_weights_kernel<<<(unsigned)(wp_elems / 256), 256, 0, stream>>>(
        weight, mask, gauge, wp);
  }
  {  // K2: banded WMMA GEMM
    dim3 grid(NTOK / 128, GROUPS);                            // (64, 256)
    banded_gemm_kernel<<<grid, 128, 0, stream>>>(xbf, wp, bias, out);
  }
}
